// SNNForecaster_34411277975782
// MI455X (gfx1250) — matrix operand order repaired and checked by reading, not compiled, not run on hardware
//
#include <hip/hip_runtime.h>
#include <hip/hip_bf16.h>

// ---------------------------------------------------------------------------
// Problem constants (from reference): B=128, T=1024, F=32, H=256, HOR=96
// ---------------------------------------------------------------------------
#define BSZ   128
#define TLEN  1024
#define TM1   1023
#define FDIM  32
#define K2F   64          // 2*F
#define HDIM  256
#define HHALF 128         // H/2 (attn hidden)
#define HOR   96
#define NROWS (BSZ * TM1) // 130944 flattened (b,t) rows

typedef __attribute__((ext_vector_type(16))) __bf16 v16bf;
typedef __attribute__((ext_vector_type(8)))  float  v8f;

__device__ __forceinline__ __bf16 f2bf(float f) { return (__bf16)f; }

// ---------------------------------------------------------------------------
// Kernel 0: zero the per-channel stat accumulators
// ---------------------------------------------------------------------------
__global__ __launch_bounds__(256) void zero_stats_kernel(float* __restrict__ g_sum,
                                                         float* __restrict__ g_sumsq) {
  g_sum[threadIdx.x]   = 0.0f;
  g_sumsq[threadIdx.x] = 0.0f;
}

// ---------------------------------------------------------------------------
// Kernel 1: projection GEMM  h = combined @ proj_w^T + proj_b
//   combined[r, k] = k<32 ? x[b,t+1,k] : x[b,t+1,k-32]-x[b,t,k-32],  r=b*1023+t
//   M = NROWS (tile 16 per block), N = 256 (4 waves x 4 ntiles), K = 64 (2 steps)
//   bf16 WMMA, f32 accumulate.
// ---------------------------------------------------------------------------
__global__ __launch_bounds__(128) void proj_gemm_kernel(
    const float* __restrict__ x, const float* __restrict__ proj_w,
    const float* __restrict__ proj_b, float* __restrict__ h) {
  const int lane = threadIdx.x & 31;
  const int wave = threadIdx.x >> 5;
  const int half = lane >> 4;   // 0: lanes 0-15, 1: lanes 16-31
  const int l16  = lane & 15;

  const long rowBase = (long)blockIdx.x * 16;
  const int  r = (int)rowBase + l16;      // A-fragment row for this lane
  const int  b = r / TM1;
  const int  t = r % TM1;
  const float* xr1 = x + (long)(b * TLEN + t + 1) * FDIM;  // x[b, t+1, :]
  const float* xr0 = x + (long)(b * TLEN + t    ) * FDIM;  // x[b, t,   :]

  // A fragments (16x32 bf16, wave32 layout): element j -> K_local
  //   j   : K = half*16 + j         (K 0-15 / 16-31)
  v16bf a0, a1;
#pragma unroll
  for (int j = 0; j < 16; ++j) {
    const int kl = half * 16 + j;
    a0[j] = f2bf(xr1[kl]);              // ktile 0: raw x[:,1:,:]
    a1[j] = f2bf(xr1[kl] - xr0[kl]);    // ktile 1: delta
  }

  const int nColBase = wave * 64;       // each wave owns 64 output channels
#pragma unroll
  for (int nt = 0; nt < 4; ++nt) {
    const int ncol = nColBase + nt * 16 + l16;
    // B fragments (32x16 bf16): lane = column, element j -> K = half*16 + j
    const float* wr = proj_w + (long)ncol * K2F;
    v16bf b0, b1;
#pragma unroll
    for (int j = 0; j < 16; ++j) {
      const int kl = half * 16 + j;
      b0[j] = f2bf(wr[kl]);
      b1[j] = f2bf(wr[32 + kl]);
    }
    v8f c = {};
    c = __builtin_amdgcn_wmma_f32_16x16x32_bf16(false, a0, false, b0, (short)0, c, false, false);
    c = __builtin_amdgcn_wmma_f32_16x16x32_bf16(false, a1, false, b1, (short)0, c, false, false);

    const float bias = proj_b[ncol];
#pragma unroll
    for (int i = 0; i < 8; ++i) {
      const long gr = rowBase + i + half * 8;   // D layout: M = i + 8*half
      h[gr * HDIM + ncol] = c[i] + bias;
    }
  }
}

// ---------------------------------------------------------------------------
// Kernel 2: per-channel sum / sumsq over all (b,t) rows of h
// ---------------------------------------------------------------------------
__global__ __launch_bounds__(256) void stats_kernel(const float* __restrict__ h,
                                                    float* __restrict__ g_sum,
                                                    float* __restrict__ g_sumsq) {
  const int  c  = threadIdx.x;
  const long r0 = (long)blockIdx.x * 128;
  const float* p = h + r0 * HDIM + c;
  float s = 0.0f, s2 = 0.0f;
#pragma unroll 4
  for (int i = 0; i < 128; ++i) {
    const float v = p[(long)i * HDIM];
    s += v; s2 += v * v;
  }
  atomicAdd(&g_sum[c], s);
  atomicAdd(&g_sumsq[c], s2);
}

// ---------------------------------------------------------------------------
// Kernel 3: finalize BN -> fused scale/shift  (h_norm = h*scale + shift)
// ---------------------------------------------------------------------------
__global__ __launch_bounds__(256) void bn_finalize_kernel(
    const float* __restrict__ g_sum, const float* __restrict__ g_sumsq,
    const float* __restrict__ bn_g, const float* __restrict__ bn_b,
    float* __restrict__ g_scale, float* __restrict__ g_shift) {
  const int   c    = threadIdx.x;
  const float n    = (float)NROWS;
  const float mean = g_sum[c] / n;
  const float var  = g_sumsq[c] / n - mean * mean;
  const float sc   = bn_g[c] * rsqrtf(var + 1e-5f);
  g_scale[c] = sc;
  g_shift[c] = bn_b[c] - mean * sc;
}

// ---------------------------------------------------------------------------
// Kernel 4: 3-layer LIF scan over time (persistent state in registers)
//   one thread per (b, c); coalesced row reads/writes; sequential over t.
// ---------------------------------------------------------------------------
__global__ __launch_bounds__(256) void lif_scan_kernel(
    const float* __restrict__ h, const float* __restrict__ g_scale,
    const float* __restrict__ g_shift, float* __restrict__ mem_rec) {
  const int b = blockIdx.x;
  const int c = threadIdx.x;
  const float sc = g_scale[c];
  const float sh = g_shift[c];
  const float* hp = h       + (long)b * TM1 * HDIM + c;
  float*       mp = mem_rec + (long)b * TM1 * HDIM + c;

  float m1 = 0.0f, m2 = 0.0f, m3 = 0.0f;
  for (int t = 0; t < TM1; ++t) {
    __builtin_prefetch(hp + (long)(t + 8) * HDIM, 0, 1);
    const float inp = hp[(long)t * HDIM] * sc + sh;
    // reset from PREVIOUS membrane (subtract), spike from NEW membrane
    const float r1 = (m1 > 1.0f) ? 1.0f : 0.0f;
    m1 = 0.9f * m1 + inp - r1;
    const float s1 = (m1 > 1.0f) ? 1.0f : 0.0f;
    const float r2 = (m2 > 1.0f) ? 1.0f : 0.0f;
    m2 = 0.9f * m2 + s1 - r2;
    const float s2 = (m2 > 1.0f) ? 1.0f : 0.0f;
    const float r3 = (m3 > 1.0f) ? 1.0f : 0.0f;
    m3 = 0.9f * m3 + s2 - r3;
    mp[(long)t * HDIM] = m3;
  }
}

// ---------------------------------------------------------------------------
// Kernel 5: attention scores  a[r] = relu(mem_rec[r,:] @ W1^T + b1) . w2 + b2
//   M tile 16 per wave (4 waves/block), N = 128 (8 ntiles), K = 256 (8 ksteps)
//   bf16 WMMA + in-register relu/dot + 16-lane shfl_xor column reduction.
// ---------------------------------------------------------------------------
__global__ __launch_bounds__(128) void attn_score_kernel(
    const float* __restrict__ mem_rec, const float* __restrict__ attn_w1,
    const float* __restrict__ attn_b1, const float* __restrict__ attn_w2,
    const float* __restrict__ attn_b2, float* __restrict__ a_out) {
  const int lane = threadIdx.x & 31;
  const int wave = threadIdx.x >> 5;
  const int half = lane >> 4;
  const int l16  = lane & 15;

  const long rowBase = ((long)blockIdx.x * 4 + wave) * 16;
  const float* mrow = mem_rec + (rowBase + l16) * HDIM;

  v8f acc[8] = {};
#pragma unroll
  for (int kt = 0; kt < 8; ++kt) {
    v16bf a;
#pragma unroll
    for (int j = 0; j < 16; ++j) {
      const int kl = half * 16 + j;
      a[j] = f2bf(mrow[kt * 32 + kl]);
    }
#pragma unroll
    for (int nt = 0; nt < 8; ++nt) {
      const int ncol = nt * 16 + l16;
      const float* wr = attn_w1 + (long)ncol * HDIM + kt * 32 + half * 16;
      v16bf bb;
#pragma unroll
      for (int j = 0; j < 16; ++j) bb[j] = f2bf(wr[j]);
      acc[nt] = __builtin_amdgcn_wmma_f32_16x16x32_bf16(false, a, false, bb,
                                                        (short)0, acc[nt], false, false);
    }
  }

  // relu + contract N=128 with w2: per-lane partials over its column strip
  float rp[8] = {0, 0, 0, 0, 0, 0, 0, 0};
#pragma unroll
  for (int nt = 0; nt < 8; ++nt) {
    const int col = nt * 16 + l16;
    const float b1v = attn_b1[col];
    const float w2v = attn_w2[col];
#pragma unroll
    for (int i = 0; i < 8; ++i) {
      float v = acc[nt][i] + b1v;
      v = v > 0.0f ? v : 0.0f;
      rp[i] += v * w2v;
    }
  }
  // reduce over the 16 lanes of each half (rows differ per half, not per lane)
#pragma unroll
  for (int off = 1; off < 16; off <<= 1) {
#pragma unroll
    for (int i = 0; i < 8; ++i) rp[i] += __shfl_xor(rp[i], off, 32);
  }
  if (l16 == 0) {
    const float b2v = attn_b2[0];
#pragma unroll
    for (int i = 0; i < 8; ++i) a_out[rowBase + half * 8 + i] = rp[i] + b2v;
  }
}

// ---------------------------------------------------------------------------
// Kernel 6: softmax over time + weighted pooling + head GEMV (one block per b)
// ---------------------------------------------------------------------------
__global__ __launch_bounds__(256) void pool_head_kernel(
    const float* __restrict__ a_s, const float* __restrict__ mem_rec,
    const float* __restrict__ head_w, const float* __restrict__ head_b,
    float* __restrict__ out) {
  __shared__ float p[TM1 + 1];
  __shared__ float red[256];
  __shared__ float feat[HDIM];

  const int b = blockIdx.x;
  const int c = threadIdx.x;
  const float* ap = a_s + (long)b * TM1;

  // 1) max over time
  float lm = -3.4e38f;
  for (int t = c; t < TM1; t += 256) lm = fmaxf(lm, ap[t]);
  red[c] = lm; __syncthreads();
  for (int s = 128; s > 0; s >>= 1) {
    if (c < s) red[c] = fmaxf(red[c], red[c + s]);
    __syncthreads();
  }
  const float mx = red[0];
  __syncthreads();

  // 2) exp + sum
  float ls = 0.0f;
  for (int t = c; t < TM1; t += 256) { const float e = expf(ap[t] - mx); p[t] = e; ls += e; }
  red[c] = ls; __syncthreads();
  for (int s = 128; s > 0; s >>= 1) {
    if (c < s) red[c] += red[c + s];
    __syncthreads();
  }
  const float inv = 1.0f / red[0];
  __syncthreads();
  for (int t = c; t < TM1; t += 256) p[t] *= inv;
  __syncthreads();

  // 3) weighted temporal pooling: feat[c] = sum_t p[t] * mem_rec[b,t,c]
  const float* mp = mem_rec + (long)b * TM1 * HDIM + c;
  float acc = 0.0f;
  for (int t = 0; t < TM1; ++t) {
    __builtin_prefetch(mp + (long)(t + 8) * HDIM, 0, 1);
    acc += p[t] * mp[(long)t * HDIM];
  }
  feat[c] = acc;
  __syncthreads();

  // 4) head: out[b,o] = feat . head_w[o,:] + head_b[o]
  if (c < HOR) {
    const float* wr = head_w + (long)c * HDIM;
    float s = head_b[c];
#pragma unroll 4
    for (int k = 0; k < HDIM; ++k) s += feat[k] * wr[k];
    out[(long)b * HOR + c] = s;
  }
}

// ---------------------------------------------------------------------------
// Launch
// ---------------------------------------------------------------------------
extern "C" void kernel_launch(void* const* d_in, const int* in_sizes, int n_in,
                              void* d_out, int out_size, void* d_ws, size_t ws_size,
                              hipStream_t stream) {
  (void)in_sizes; (void)n_in; (void)out_size; (void)ws_size;
  const float* x       = (const float*)d_in[0];
  const float* proj_w  = (const float*)d_in[1];
  const float* proj_b  = (const float*)d_in[2];
  const float* bn_g    = (const float*)d_in[3];
  const float* bn_b    = (const float*)d_in[4];
  const float* attn_w1 = (const float*)d_in[5];
  const float* attn_b1 = (const float*)d_in[6];
  const float* attn_w2 = (const float*)d_in[7];
  const float* attn_b2 = (const float*)d_in[8];
  const float* head_w  = (const float*)d_in[9];
  const float* head_b  = (const float*)d_in[10];
  float* out = (float*)d_out;

  // workspace layout (256B aligned):
  //   h        : NROWS*H f32  (134.1 MB)
  //   mem_rec  : NROWS*H f32  (134.1 MB)
  //   a_s      : NROWS   f32  (0.5 MB)
  //   stats    : 4 x 256 f32
  char* ws = (char*)d_ws;
  const size_t HB = (size_t)NROWS * HDIM * sizeof(float);
  const size_t AB = (((size_t)NROWS * sizeof(float)) + 255) & ~(size_t)255;
  float* h       = (float*)ws;
  float* mem_rec = (float*)(ws + HB);
  float* a_s     = (float*)(ws + 2 * HB);
  float* g_sum   = (float*)(ws + 2 * HB + AB);
  float* g_sumsq = g_sum + HDIM;
  float* g_scale = g_sum + 2 * HDIM;
  float* g_shift = g_sum + 3 * HDIM;

  zero_stats_kernel<<<1, 256, 0, stream>>>(g_sum, g_sumsq);
  proj_gemm_kernel<<<NROWS / 16, 128, 0, stream>>>(x, proj_w, proj_b, h);
  stats_kernel<<<NROWS / 128, 256, 0, stream>>>(h, g_sum, g_sumsq);
  bn_finalize_kernel<<<1, 256, 0, stream>>>(g_sum, g_sumsq, bn_g, bn_b, g_scale, g_shift);
  lif_scan_kernel<<<BSZ, 256, 0, stream>>>(h, g_scale, g_shift, mem_rec);
  attn_score_kernel<<<NROWS / 64, 128, 0, stream>>>(mem_rec, attn_w1, attn_b1,
                                                    attn_w2, attn_b2, a_s);
  pool_head_kernel<<<BSZ, 256, 0, stream>>>(a_s, mem_rec, head_w, head_b, out);
}
